// Multiheadselfattention_7103875907922
// MI455X (gfx1250) — matrix operand order repaired and checked by reading, not compiled, not run on hardware
//
#include <hip/hip_runtime.h>
#include <hip/hip_bf16.h>
#include <math.h>

#define D_MODEL   2048
#define NUM_HEADS 16
#define HEAD_DIM  128
#define SEQ_LEN   2048
#define BATCH     4
#define NROWS     (BATCH * SEQ_LEN)   // 8192

typedef __attribute__((ext_vector_type(16))) __bf16 v16bf;
typedef __attribute__((ext_vector_type(4)))  __bf16 v4bf;
typedef __attribute__((ext_vector_type(8)))  float  v8f;

// ---------------------------------------------------------------------------
// WMMA helper: D = A(16x32 bf16) * B(32x16 bf16) + C(16x16 f32)
// ---------------------------------------------------------------------------
static __device__ __forceinline__ v8f wmma_bf16(v8f c, v16bf a, v16bf b) {
  return __builtin_amdgcn_wmma_f32_16x16x32_bf16(false, a, false, b,
                                                 (short)0, c, false, false);
}

// A-fragment loader (16x32 tile, row-major bf16 in LDS, leading dim = ld).
// lane L<16 holds row L, K = 0..15;
// lane L+16 holds row L, K = 16..31.
static __device__ __forceinline__ v16bf load_a(const __bf16* tile, int ld, int lane) {
  const int r = lane & 15, half = lane >> 4;
  const __bf16* row = tile + r * ld + half * 16;
  v16bf a;
#pragma unroll
  for (int i = 0; i < 8; ++i) a[i] = row[i];
#pragma unroll
  for (int i = 0; i < 8; ++i) a[8 + i] = row[8 + i];
  return a;
}

// B-fragment loader (32x16 tile). tile points at a [16][k] row-major block
// where row n is B-column n (a row of W, contiguous along K).
// lanes 0-15 hold K=0..15, lanes 16-31 hold K=16..31, col N = lane%16.
static __device__ __forceinline__ v16bf load_b(const __bf16* tile, int ld, int lane) {
  const int r = lane & 15, half = lane >> 4;
  const __bf16* row = tile + r * ld + half * 16;
  v16bf b;
#pragma unroll
  for (int i = 0; i < 16; ++i) b[i] = row[i];
  return b;
}

// B-fragment loader for K-major storage: tile[k][n], leading dim ld,
// column n = lane%16 (strided). Used for P*V where V is stored [key][hd].
static __device__ __forceinline__ v16bf load_b_strided(const __bf16* tile, int ld, int lane) {
  const int r = lane & 15, half = lane >> 4;
  const __bf16* p = tile + (half * 16) * ld + r;
  v16bf b;
#pragma unroll
  for (int i = 0; i < 16; ++i) b[i] = p[i * ld];
  return b;
}

// ---------------------------------------------------------------------------
// One-shot fp32 -> planar bf16 hi/lo split (hi = bf16(x), lo = bf16(x - hi)).
// Moves the conversion VALU work out of the GEMM inner loop.
// ---------------------------------------------------------------------------
__global__ void __launch_bounds__(256)
cvt_hilo_kernel(const float* __restrict__ src, __bf16* __restrict__ hi,
                __bf16* __restrict__ lo, const int n4) {
  const int i = blockIdx.x * 256 + threadIdx.x;
  if (i >= n4) return;
  const float4 f = ((const float4*)src)[i];
  v4bf h, l;
  h[0] = (__bf16)f.x; l[0] = (__bf16)(f.x - (float)h[0]);
  h[1] = (__bf16)f.y; l[1] = (__bf16)(f.y - (float)h[1]);
  h[2] = (__bf16)f.z; l[2] = (__bf16)(f.z - (float)h[2]);
  h[3] = (__bf16)f.w; l[3] = (__bf16)(f.w - (float)h[3]);
  ((v4bf*)hi)[i] = h;
  ((v4bf*)lo)[i] = l;
}

// ---------------------------------------------------------------------------
// Projection GEMM on pre-split bf16 planes:
//   C[m][n] = sum_k X[m][k] * W[n][k], split-bf16 x3 WMMA (~fp32 accuracy).
// Block tile 128x128, BK=32, 256 threads = 8 waves (4M x 2N); each wave owns
// a 32x64 quadrant = 2x4 WMMA tiles -> 24 WMMA per wave per k-step.
// mode 0: RoPE epilogue -> (B,H,S,hd)      (Q and K)
// mode 2: transpose epilogue -> (B,H,S,hd) (V)
// mode 3: plain row-major (B,S,D)          (output projection)
// ---------------------------------------------------------------------------
__global__ void __launch_bounds__(256)
proj_gemm_kernel(const __bf16* __restrict__ Xh, const __bf16* __restrict__ Xl,
                 const __bf16* __restrict__ Wh, const __bf16* __restrict__ Wl,
                 float* __restrict__ dst, const int* __restrict__ start_pos_p,
                 const int mode) {
  // pad to 36 (stride 18 dwords): rows 8B-aligned for b64 stores, and the 16
  // rows of a fragment map to distinct banks (18r mod 64 distinct, r<16).
  __shared__ __bf16 Ah[128][36];
  __shared__ __bf16 Al[128][36];
  __shared__ __bf16 Bh[128][36];
  __shared__ __bf16 Bl[128][36];

  const int tid  = threadIdx.x;
  const int lane = tid & 31;
  const int wave = tid >> 5;
  const int r    = lane & 15;
  const int half = lane >> 4;
  const int wm   = wave & 3;   // M offset wm*32
  const int wn   = wave >> 2;  // N offset wn*64
  const int m0   = blockIdx.y * 128;
  const int n0   = blockIdx.x * 128;

  const int row  = tid >> 1;         // 0..127 (shared by A and B staging)
  const int koff = (tid & 1) * 16;   // 16 bf16 per thread per plane

  const v8f z8 = {0.f, 0.f, 0.f, 0.f, 0.f, 0.f, 0.f, 0.f};
  v8f acc[2][4];
#pragma unroll
  for (int mt = 0; mt < 2; ++mt)
#pragma unroll
    for (int nt = 0; nt < 4; ++nt) acc[mt][nt] = z8;

  const size_t abase = (size_t)(m0 + row) * D_MODEL + koff;
  const size_t bbase = (size_t)(n0 + row) * D_MODEL + koff;

  for (int kt = 0; kt < D_MODEL / 32; ++kt) {
    const int kg = kt * 32;
    {  // stage A (X hi/lo) and B (W hi/lo) tiles: pure bf16 copies
      const uint4 xh0 = *(const uint4*)(Xh + abase + kg);
      const uint4 xh1 = *(const uint4*)(Xh + abase + kg + 8);
      const uint4 xl0 = *(const uint4*)(Xl + abase + kg);
      const uint4 xl1 = *(const uint4*)(Xl + abase + kg + 8);
      const uint4 wh0 = *(const uint4*)(Wh + bbase + kg);
      const uint4 wh1 = *(const uint4*)(Wh + bbase + kg + 8);
      const uint4 wl0 = *(const uint4*)(Wl + bbase + kg);
      const uint4 wl1 = *(const uint4*)(Wl + bbase + kg + 8);
      if (kt + 1 < D_MODEL / 32) {  // prefetch next k-tile
        __builtin_prefetch(Xh + abase + kg + 32, 0, 1);
        __builtin_prefetch(Xl + abase + kg + 32, 0, 1);
        __builtin_prefetch(Wh + bbase + kg + 32, 0, 1);
        __builtin_prefetch(Wl + bbase + kg + 32, 0, 1);
      }
      unsigned int* dah = (unsigned int*)&Ah[row][koff];
      unsigned int* dal = (unsigned int*)&Al[row][koff];
      unsigned int* dbh = (unsigned int*)&Bh[row][koff];
      unsigned int* dbl = (unsigned int*)&Bl[row][koff];
      dah[0] = xh0.x; dah[1] = xh0.y; dah[2] = xh0.z; dah[3] = xh0.w;
      dah[4] = xh1.x; dah[5] = xh1.y; dah[6] = xh1.z; dah[7] = xh1.w;
      dal[0] = xl0.x; dal[1] = xl0.y; dal[2] = xl0.z; dal[3] = xl0.w;
      dal[4] = xl1.x; dal[5] = xl1.y; dal[6] = xl1.z; dal[7] = xl1.w;
      dbh[0] = wh0.x; dbh[1] = wh0.y; dbh[2] = wh0.z; dbh[3] = wh0.w;
      dbh[4] = wh1.x; dbh[5] = wh1.y; dbh[6] = wh1.z; dbh[7] = wh1.w;
      dbl[0] = wl0.x; dbl[1] = wl0.y; dbl[2] = wl0.z; dbl[3] = wl0.w;
      dbl[4] = wl1.x; dbl[5] = wl1.y; dbl[6] = wl1.z; dbl[7] = wl1.w;
    }
    __syncthreads();

    v16bf ah[2], al[2];
#pragma unroll
    for (int mt = 0; mt < 2; ++mt) {
      ah[mt] = load_a(&Ah[wm * 32 + mt * 16][0], 36, lane);
      al[mt] = load_a(&Al[wm * 32 + mt * 16][0], 36, lane);
    }
#pragma unroll
    for (int nt = 0; nt < 4; ++nt) {
      const v16bf bh = load_b(&Bh[wn * 64 + nt * 16][0], 36, lane);
      const v16bf bl = load_b(&Bl[wn * 64 + nt * 16][0], 36, lane);
#pragma unroll
      for (int mt = 0; mt < 2; ++mt) {
        acc[mt][nt] = wmma_bf16(acc[mt][nt], ah[mt], bh);  // hi*hi
        acc[mt][nt] = wmma_bf16(acc[mt][nt], al[mt], bh);  // lo*hi
        acc[mt][nt] = wmma_bf16(acc[mt][nt], ah[mt], bl);  // hi*lo
      }
    }
    __syncthreads();
  }

  const int sp = start_pos_p[0];

  // Epilogue. C-fragment layout: element i -> row = i + 8*half, col = r.
#pragma unroll
  for (int mt = 0; mt < 2; ++mt) {
#pragma unroll
    for (int nt = 0; nt < 4; ++nt) {
#pragma unroll
      for (int i = 0; i < 8; ++i) {
        const int m   = m0 + wm * 32 + mt * 16 + i + 8 * half;
        const int col = n0 + wn * 64 + nt * 16 + r;
        const float f = acc[mt][nt][i];
        if (mode == 3) {
          dst[(size_t)m * D_MODEL + col] = f;
        } else {
          const int bb = m / SEQ_LEN;
          const int ss = m % SEQ_LEN;
          if (mode == 2) {  // V: transpose to (B,H,S,hd)
            const int hh = col >> 7, idx = col & 127;
            dst[(((size_t)bb * NUM_HEADS + hh) * SEQ_LEN + ss) * HEAD_DIM + idx] = f;
          } else {          // Q/K: fused RoPE, then (B,H,S,hd)
            // pair (2d,2d+1) lives in adjacent lanes (col parity == r parity)
            const float partner = __shfl_xor(f, 1, 32);
            const int d = col >> 1;
            const float theta =
                expf((float)d * (-2.0f / (float)D_MODEL) * 9.210340371976184f);
            const float ang = (float)(sp + ss) * theta;
            float sn, cs;
            sincosf(ang, &sn, &cs);
            float val; int dim;
            if ((r & 1) == 0) { val = f * cs - partner * sn; dim = d; }
            else              { val = partner * sn + f * cs; dim = d + D_MODEL / 2; }
            const int hh = dim >> 7, idx = dim & 127;
            dst[(((size_t)bb * NUM_HEADS + hh) * SEQ_LEN + ss) * HEAD_DIM + idx] = val;
          }
        }
      }
    }
  }
}

// ---------------------------------------------------------------------------
// Causal flash attention. One block = 4 waves x 16 query rows (BQ=64),
// key tiles of 64, online softmax, bf16 WMMA for QK^T and P*V.
// q,k,v in (B,H,S,hd); output to (B,S,D) scratch for the final projection.
// ---------------------------------------------------------------------------
__global__ void __launch_bounds__(128)
attn_kernel(const float* __restrict__ q, const float* __restrict__ k,
            const float* __restrict__ v, float* __restrict__ outp) {
  __shared__ __bf16 Qb[64][132];
  __shared__ __bf16 Kb[64][132];
  __shared__ __bf16 Vb[64][132];
  __shared__ __bf16 Pst[4][16][68];

  const int qt   = blockIdx.x;
  const int bh   = blockIdx.y;
  const int bb   = bh >> 4;
  const int hh   = bh & 15;
  const int tid  = threadIdx.x;
  const int wave = tid >> 5;
  const int lane = tid & 31;
  const int r    = lane & 15;
  const int half = lane >> 4;
  const int q0   = qt * 64;
  const float scale = 0.08838834764831845f;  // 1/sqrt(128)

  const float* qb = q + (size_t)bh * SEQ_LEN * HEAD_DIM;
  const float* kb = k + (size_t)bh * SEQ_LEN * HEAD_DIM;
  const float* vb = v + (size_t)bh * SEQ_LEN * HEAD_DIM;

  // load Q tile (64 x 128) -> bf16 LDS
  for (int f4i = tid; f4i < 64 * HEAD_DIM / 4; f4i += 128) {
    const int row = f4i >> 5, c = (f4i & 31) * 4;
    const float4 t = *(const float4*)(qb + (size_t)(q0 + row) * HEAD_DIM + c);
    Qb[row][c + 0] = (__bf16)t.x; Qb[row][c + 1] = (__bf16)t.y;
    Qb[row][c + 2] = (__bf16)t.z; Qb[row][c + 3] = (__bf16)t.w;
  }

  const v8f z8 = {0.f, 0.f, 0.f, 0.f, 0.f, 0.f, 0.f, 0.f};
  v8f O[8];
#pragma unroll
  for (int j = 0; j < 8; ++j) O[j] = z8;
  float mrun[8], lrun[8];
#pragma unroll
  for (int i = 0; i < 8; ++i) { mrun[i] = -3.0e38f; lrun[i] = 0.f; }

  const int nkt = qt + 1;  // causal: skip fully-masked key tiles
  for (int kt = 0; kt < nkt; ++kt) {
    const int k0 = kt * 64;
    __syncthreads();  // protect Kb/Vb (and Qb on first pass)
    for (int f4i = tid; f4i < 64 * HEAD_DIM / 4; f4i += 128) {
      const int row = f4i >> 5, c = (f4i & 31) * 4;
      const float4 tk = *(const float4*)(kb + (size_t)(k0 + row) * HEAD_DIM + c);
      const float4 tv = *(const float4*)(vb + (size_t)(k0 + row) * HEAD_DIM + c);
      Kb[row][c + 0] = (__bf16)tk.x; Kb[row][c + 1] = (__bf16)tk.y;
      Kb[row][c + 2] = (__bf16)tk.z; Kb[row][c + 3] = (__bf16)tk.w;
      Vb[row][c + 0] = (__bf16)tv.x; Vb[row][c + 1] = (__bf16)tv.y;
      Vb[row][c + 2] = (__bf16)tv.z; Vb[row][c + 3] = (__bf16)tv.w;
    }
    __syncthreads();

    // scores: S(16x64) = Q(16x128) * K^T ; 4 hd-steps of 32
    v16bf af[4];
#pragma unroll
    for (int ks = 0; ks < 4; ++ks)
      af[ks] = load_a(&Qb[wave * 16][0] + ks * 32, 132, lane);
    v8f sc[4];
#pragma unroll
    for (int nt = 0; nt < 4; ++nt) {
      sc[nt] = z8;
#pragma unroll
      for (int ks = 0; ks < 4; ++ks) {
        const v16bf bf = load_b(&Kb[nt * 16][0] + ks * 32, 132, lane);
        sc[nt] = wmma_bf16(sc[nt], af[ks], bf);
      }
    }

    // scale + causal mask + online softmax
    float val[4][8], mloc[8];
#pragma unroll
    for (int i = 0; i < 8; ++i) mloc[i] = -3.0e38f;
#pragma unroll
    for (int nt = 0; nt < 4; ++nt)
#pragma unroll
      for (int i = 0; i < 8; ++i) {
        const int qrow = q0 + wave * 16 + i + 8 * half;
        const int kcol = k0 + nt * 16 + r;
        float x = sc[nt][i] * scale;
        if (kcol > qrow) x = -3.0e38f;
        val[nt][i] = x;
        mloc[i] = fmaxf(mloc[i], x);
      }
#pragma unroll
    for (int i = 0; i < 8; ++i)
      for (int off = 1; off < 16; off <<= 1)
        mloc[i] = fmaxf(mloc[i], __shfl_xor(mloc[i], off, 32));
    float corr[8], ps[8];
#pragma unroll
    for (int i = 0; i < 8; ++i) {
      const float mnew = fmaxf(mrun[i], mloc[i]);
      corr[i] = __expf(mrun[i] - mnew);
      mrun[i] = mnew;
      ps[i] = 0.f;
    }
#pragma unroll
    for (int nt = 0; nt < 4; ++nt)
#pragma unroll
      for (int i = 0; i < 8; ++i) {
        const float p = __expf(val[nt][i] - mrun[i]);
        Pst[wave][i + 8 * half][nt * 16 + r] = (__bf16)p;
        ps[i] += p;
      }
#pragma unroll
    for (int i = 0; i < 8; ++i) {
      for (int off = 1; off < 16; off <<= 1) ps[i] += __shfl_xor(ps[i], off, 32);
      lrun[i] = lrun[i] * corr[i] + ps[i];
    }
#pragma unroll
    for (int j = 0; j < 8; ++j)
#pragma unroll
      for (int i = 0; i < 8; ++i) O[j][i] *= corr[i];
    __syncthreads();  // Pst cross-lane visibility before A-fragment reads

    // O(16x128) += P(16x64) * V(64x128); 2 key-steps of 32, 8 hd-tiles
#pragma unroll
    for (int ks2 = 0; ks2 < 2; ++ks2) {
      const v16bf ap = load_a(&Pst[wave][0][0] + ks2 * 32, 68, lane);
#pragma unroll
      for (int j = 0; j < 8; ++j) {
        const v16bf bv2 = load_b_strided(&Vb[ks2 * 32][0] + j * 16, 132, lane);
        O[j] = wmma_bf16(O[j], ap, bv2);
      }
    }
  }

  // normalize + write (B,S,D)
#pragma unroll
  for (int i = 0; i < 8; ++i) {
    const float inv = 1.0f / lrun[i];
    const int qrow = q0 + wave * 16 + i + 8 * half;
#pragma unroll
    for (int j = 0; j < 8; ++j)
      outp[(size_t)(bb * SEQ_LEN + qrow) * D_MODEL + hh * HEAD_DIM + j * 16 + r] =
          O[j][i] * inv;
  }
}

// ---------------------------------------------------------------------------
// inputs: x, w_q, w_k, w_v, w_o, start_pos
// d_out: [output (B,S,D)] [k (B,H,S,hd)] [v (B,H,S,hd)]
// d_ws : qrope(64M) | attno(64M) | Xh(32M) | Xl(32M) | 4x (Wh(8M) | Wl(8M))
//        = 256 MiB. attno's bf16 planes reuse Xh/Xl (dead after QKV GEMMs).
// ---------------------------------------------------------------------------
extern "C" void kernel_launch(void* const* d_in, const int* in_sizes, int n_in,
                              void* d_out, int out_size, void* d_ws, size_t ws_size,
                              hipStream_t stream) {
  const float* x  = (const float*)d_in[0];
  const float* w[4] = {(const float*)d_in[1], (const float*)d_in[2],
                       (const float*)d_in[3], (const float*)d_in[4]};
  const int* sp = (const int*)d_in[5];

  const size_t tensor_elems = (size_t)BATCH * SEQ_LEN * D_MODEL;  // 16.7M
  const size_t w_elems = (size_t)D_MODEL * D_MODEL;               // 4.19M

  float* out  = (float*)d_out;
  float* kout = out + tensor_elems;
  float* vout = kout + tensor_elems;

  char* ws = (char*)d_ws;
  float* qrope = (float*)ws;           ws += tensor_elems * 4;
  float* attno = (float*)ws;           ws += tensor_elems * 4;
  __bf16* Xh = (__bf16*)ws;            ws += tensor_elems * 2;
  __bf16* Xl = (__bf16*)ws;            ws += tensor_elems * 2;
  __bf16 *Whp[4], *Wlp[4];
  for (int i = 0; i < 4; ++i) {
    Whp[i] = (__bf16*)ws;              ws += w_elems * 2;
    Wlp[i] = (__bf16*)ws;              ws += w_elems * 2;
  }

  // one-shot hi/lo splits
  cvt_hilo_kernel<<<dim3((unsigned)(tensor_elems / 4 / 256)), dim3(256), 0, stream>>>(
      x, Xh, Xl, (int)(tensor_elems / 4));
  for (int i = 0; i < 4; ++i)
    cvt_hilo_kernel<<<dim3((unsigned)(w_elems / 4 / 256)), dim3(256), 0, stream>>>(
        w[i], Whp[i], Wlp[i], (int)(w_elems / 4));

  const dim3 gblk(256);
  const dim3 ggrd(D_MODEL / 128, NROWS / 128);  // 16 x 64

  proj_gemm_kernel<<<ggrd, gblk, 0, stream>>>(Xh, Xl, Whp[0], Wlp[0], qrope, sp, 0);
  proj_gemm_kernel<<<ggrd, gblk, 0, stream>>>(Xh, Xl, Whp[1], Wlp[1], kout,  sp, 0);
  proj_gemm_kernel<<<ggrd, gblk, 0, stream>>>(Xh, Xl, Whp[2], Wlp[2], vout,  sp, 2);

  attn_kernel<<<dim3(SEQ_LEN / 64, BATCH * NUM_HEADS), dim3(128), 0, stream>>>(
      qrope, kout, vout, attno);

  // reuse Xh/Xl planes for the attention output's hi/lo split
  cvt_hilo_kernel<<<dim3((unsigned)(tensor_elems / 4 / 256)), dim3(256), 0, stream>>>(
      attno, Xh, Xl, (int)(tensor_elems / 4));
  proj_gemm_kernel<<<ggrd, gblk, 0, stream>>>(Xh, Xl, Whp[3], Wlp[3], out, sp, 3);
}
